// ScaledDotProduct_18597208392097
// MI455X (gfx1250) — matrix operand order repaired and checked by reading, not compiled, not run on hardware
//
#include <hip/hip_runtime.h>

typedef __attribute__((ext_vector_type(16))) __bf16 bf16x16;
typedef __attribute__((ext_vector_type(8)))  float  f32x8;
typedef unsigned int u32x4 __attribute__((ext_vector_type(4)));
typedef int          i32x4 __attribute__((ext_vector_type(4)));
typedef int          i32x8 __attribute__((ext_vector_type(8)));

#ifndef __has_builtin
#define __has_builtin(x) 0
#endif

#if __has_builtin(__builtin_amdgcn_tensor_load_to_lds) && \
    __has_builtin(__builtin_amdgcn_s_wait_tensorcnt)
#define USE_TDM 1
#else
#define USE_TDM 0
#endif

#if __has_include(<hip/amd_detail/amd_gfx1250_TDM.h>)
#define TDM_SIX_ARGS 1   // amdgpu-toolchain / therock headers -> 6-arg builtin
#else
#define TDM_SIX_ARGS 0   // ROCm 7.2 -> 5-arg builtin
#endif

#define BM 128
#define BN 128
#define BK 32
#define LDT 40   // padded LDS row stride (bf16 elems): 64B row + 16B pad = 80B

union FragAB {
    bf16x16 v;
    uint4   u[2];
};

#if USE_TDM
// One 2D TDM descriptor: 128 rows x 32 bf16 (K-contiguous), row stride ld elems,
// LDS destination padded by hardware (pad_interval = 64B, pad_amount = 16B -> LDT=40).
__device__ __forceinline__ void tdm_load_tile(const __bf16* gsrc, int ld_elems,
                                              unsigned lds_byte_off)
{
    const unsigned long long ga = (unsigned long long)gsrc;
    u32x4 g0;
    g0[0] = 1u;                                           // count=1 (valid user D#)
    g0[1] = lds_byte_off;                                 // lds_addr
    g0[2] = (unsigned)(ga & 0xFFFFFFFFu);                 // global_addr[31:0]
    g0[3] = (unsigned)((ga >> 32) & 0x01FFFFFFu)          // global_addr[56:32]
          | (2u << 30);                                   // type = 2 ("image")
    i32x8 g1;
    g1[0] = (1 << 16)      // data_size = 2 bytes
          | (1 << 20)      // pad_enable
          | (3 << 22)      // pad_interval: 64B between pads (one 32-elem row)
          | (3 << 25);     // pad_amount: 4 DWORDs = 16B
    g1[1] = (int)((unsigned)BK << 16);                    // tensor_dim0[15:0]=32
    g1[2] = (int)((unsigned)BM << 16);                    // dim0 hi | tensor_dim1[15:0]=128
    g1[3] = (int)((unsigned)BK << 16);                    // dim1 hi | tile_dim0=32
    g1[4] = BM;                                           // tile_dim1=128 | tile_dim2=0
    g1[5] = ld_elems;                                     // tensor_dim0_stride[31:0]
    g1[6] = 0;
    g1[7] = 0;
    const i32x4 z4 = {};
#if TDM_SIX_ARGS
    const i32x8 z8 = {};
    __builtin_amdgcn_tensor_load_to_lds(g0, g1, z4, z4, z8, 0);
#else
    __builtin_amdgcn_tensor_load_to_lds(g0, g1, z4, z4, 0);
#endif
}
#endif // USE_TDM

// bf16 WMMA GEMM, all operands bf16, B pre-laid-out transposed:
//   A : [M x K] row-major (K contiguous), ld = lda
//   Bt: B[k][n] = Bt[n*ldb + k]   (K contiguous), ld = ldb
//   C : f32 or bf16; OUT_TRANS stores C[col*ldc + row] instead of C[row*ldc + col]
template<bool ADD_BIAS, bool OUT_BF16, bool OUT_TRANS>
__global__ __launch_bounds__(256)
void wmma_gemm(const __bf16* __restrict__ A, const __bf16* __restrict__ Bt,
               const float* __restrict__ bias, void* __restrict__ Cv,
               int K, int lda, int ldb, int ldc,
               long batchA, long batchB, long batchC, float alpha)
{
    __shared__ __align__(16) __bf16 lds[2][2][BM * LDT];  // [buf][A/B][row*LDT+k]

    const int tid    = threadIdx.x;
    const int lane   = tid & 31;
    const int wave   = tid >> 5;
    const int hiHalf = lane >> 4;
    const int l16    = lane & 15;

    const int tileM0 = blockIdx.y * BM;
    const int tileN0 = blockIdx.x * BN;
    const int z      = blockIdx.z;

    const int waveM = (wave >> 2) * 64;   // 0 / 64
    const int waveN = (wave & 3) * 32;    // 0..96

    const __bf16* Abase = A  + (long)z * batchA + (long)tileM0 * lda;
    const __bf16* Bbase = Bt + (long)z * batchB + (long)tileN0 * ldb;

    f32x8 acc[4][2] = {};

    auto mma_step = [&](int b) {
        const __bf16* tA = &lds[b][0][0];
        const __bf16* tB = &lds[b][1][0];
        FragAB afr[4], bfr[2];
        #pragma unroll
        for (int mi = 0; mi < 4; ++mi) {
            // 16-bit A 16x32: lanes 0-15: K0..15 ; lanes 16-31: K16..31
            const __bf16* base = tA + (waveM + mi * 16 + l16) * LDT;
            afr[mi].u[0] = *(const uint4*)(base + (hiHalf ? 16 : 0));
            afr[mi].u[1] = *(const uint4*)(base + (hiHalf ? 24 : 8));
        }
        #pragma unroll
        for (int ni = 0; ni < 2; ++ni) {
            // 16-bit B 32x16: col = lane%16; lanes 0-15: K0..15, lanes 16-31: K16..31
            const __bf16* base = tB + (waveN + ni * 16 + l16) * LDT + hiHalf * 16;
            bfr[ni].u[0] = *(const uint4*)(base);
            bfr[ni].u[1] = *(const uint4*)(base + 8);
        }
        #pragma unroll
        for (int mi = 0; mi < 4; ++mi)
            #pragma unroll
            for (int ni = 0; ni < 2; ++ni)
                acc[mi][ni] = __builtin_amdgcn_wmma_f32_16x16x32_bf16(
                    false, afr[mi].v, false, bfr[ni].v,
                    (short)0, acc[mi][ni], false, false);
    };

    const int nIter = K / BK;

#if USE_TDM
    auto tdm_issue = [&](int it) {
        const int b = it & 1;
        if (wave == 0)
            tdm_load_tile(Abase + (long)it * BK, lda,
                          (unsigned)(unsigned long long)&lds[b][0][0]);
        else if (wave == 1)
            tdm_load_tile(Bbase + (long)it * BK, ldb,
                          (unsigned)(unsigned long long)&lds[b][1][0]);
    };
    // Double-buffered TDM pipeline: DMA tile it+1 while WMMAs consume tile it.
    tdm_issue(0);
    for (int it = 0; it < nIter - 1; ++it) {
        tdm_issue(it + 1);
        if (wave < 2) __builtin_amdgcn_s_wait_tensorcnt(1);  // tile 'it' landed
        __syncthreads();
        mma_step(it & 1);
        __syncthreads();   // everyone done reading before DMA overwrites this buf
    }
    if (wave < 2) __builtin_amdgcn_s_wait_tensorcnt(0);
    __syncthreads();
    mma_step((nIter - 1) & 1);
#else
    auto copy_tile = [&](const __bf16* src, int ld, __bf16* dst) {
        #pragma unroll
        for (int i = 0; i < 2; ++i) {
            const int c  = tid + i * 256;    // 0..511 chunks of 8 halves
            const int r  = c >> 2;           // 0..127
            const int kq = (c & 3) * 8;      // 0,8,16,24
            *(uint4*)(dst + r * LDT + kq) = *(const uint4*)(src + (long)r * ld + kq);
        }
    };
    for (int it = 0; it < nIter; ++it) {
        const int b = it & 1;
        copy_tile(Abase + (long)it * BK, lda, &lds[b][0][0]);
        copy_tile(Bbase + (long)it * BK, ldb, &lds[b][1][0]);
        __syncthreads();
        mma_step(b);
        __syncthreads();
    }
#endif

    // ---------------- epilogue ----------------
    #pragma unroll
    for (int mi = 0; mi < 4; ++mi) {
        #pragma unroll
        for (int ni = 0; ni < 2; ++ni) {
            const int col  = tileN0 + waveN + ni * 16 + l16;
            const float bv = ADD_BIAS ? bias[col] : 0.0f;
            #pragma unroll
            for (int r = 0; r < 8; ++r) {
                const int row = tileM0 + waveM + mi * 16 + hiHalf * 8 + r;
                const float val = acc[mi][ni][r] * alpha + bv;
                const long  idx = OUT_TRANS ? (long)col * ldc + row
                                            : (long)row * ldc + col;
                if (OUT_BF16)
                    ((__bf16*)Cv + (long)z * batchC)[idx] = (__bf16)val;
                else
                    ((float*)Cv  + (long)z * batchC)[idx] = val;
            }
        }
    }
}

// fp32 -> bf16 copy-convert (n multiple of 4)
__global__ __launch_bounds__(256)
void cvt_f32_bf16(const float* __restrict__ in, __bf16* __restrict__ out, long n)
{
    const long i = ((long)blockIdx.x * 256 + threadIdx.x) * 4;
    if (i + 3 < n) {
        const float4 f = *(const float4*)(in + i);
        union { __bf16 h[4]; uint2 u; } pk;
        pk.h[0] = (__bf16)f.x; pk.h[1] = (__bf16)f.y;
        pk.h[2] = (__bf16)f.z; pk.h[3] = (__bf16)f.w;
        *(uint2*)(out + i) = pk.u;
    }
}

// fp32 [K x N] -> bf16 transposed [N x K] (32x32 LDS tiles)
__global__ __launch_bounds__(256)
void transpose_cvt(const float* __restrict__ in, __bf16* __restrict__ out, int K, int N)
{
    __shared__ float t[32][33];
    const int k0 = blockIdx.y * 32, n0 = blockIdx.x * 32;
    const int tx = threadIdx.x & 31, ty = threadIdx.x >> 5;
    #pragma unroll
    for (int i = 0; i < 32; i += 8)
        t[ty + i][tx] = in[(long)(k0 + ty + i) * N + n0 + tx];
    __syncthreads();
    #pragma unroll
    for (int i = 0; i < 32; i += 8)
        out[(long)(n0 + ty + i) * K + k0 + tx] = (__bf16)t[tx][ty + i];
}

// Row softmax over 2048 columns -> bf16
__global__ __launch_bounds__(256)
void softmax_rows(const float* __restrict__ logits, __bf16* __restrict__ attn, int ncols)
{
    __shared__ float red[256];
    const long  row = blockIdx.x;
    const float* in = logits + row * ncols;
    __bf16*     out = attn   + row * ncols;
    const int   tid = threadIdx.x;

    float x[8];
    float m = -3.4e38f;
    #pragma unroll
    for (int i = 0; i < 8; ++i) { x[i] = in[tid + i * 256]; m = fmaxf(m, x[i]); }
    red[tid] = m; __syncthreads();
    for (int s = 128; s > 0; s >>= 1) {
        if (tid < s) red[tid] = fmaxf(red[tid], red[tid + s]);
        __syncthreads();
    }
    m = red[0]; __syncthreads();

    float sum = 0.0f;
    #pragma unroll
    for (int i = 0; i < 8; ++i) { x[i] = __expf(x[i] - m); sum += x[i]; }
    red[tid] = sum; __syncthreads();
    for (int s = 128; s > 0; s >>= 1) {
        if (tid < s) red[tid] += red[tid + s];
        __syncthreads();
    }
    const float inv = 1.0f / red[0];
    #pragma unroll
    for (int i = 0; i < 8; ++i) out[tid + i * 256] = (__bf16)(x[i] * inv);
}

extern "C" void kernel_launch(void* const* d_in, const int* in_sizes, int n_in,
                              void* d_out, int out_size, void* d_ws, size_t ws_size,
                              hipStream_t stream)
{
    const float* q  = (const float*)d_in[0];
    const float* k  = (const float*)d_in[1];
    const float* v  = (const float*)d_in[2];
    const float* Wq = (const float*)d_in[3];
    const float* bq = (const float*)d_in[4];
    const float* Wk = (const float*)d_in[5];
    const float* bk = (const float*)d_in[6];
    const float* Wv = (const float*)d_in[7];
    const float* bv = (const float*)d_in[8];
    float* out = (float*)d_out;

    const int  Bn = 4, Sn = 2048, Dn = 1024;
    const int  Mp = Bn * Sn;                 // 8192
    const long Pe = (long)Mp * Dn;           // 8,388,608  activation elems
    const long We = (long)Dn * Dn;           // 1,048,576  weight elems
    const long Le = (long)Bn * Sn * Sn;      // 16,777,216 logits/attn elems

    char* ws = (char*)d_ws;
    // Phase-1 region (dead after projections) is overlaid by f32 logits (64MB >= 54MB):
    __bf16* qb  = (__bf16*)ws;
    __bf16* kb  = qb  + Pe;
    __bf16* vb  = kb  + Pe;
    __bf16* Wqt = vb  + Pe;
    __bf16* Wkt = Wqt + We;
    __bf16* Wvt = Wkt + We;
    float*  Lg  = (float*)ws;
    char*   p2  = ws + Le * 4;
    __bf16* Qp  = (__bf16*)p2;               // [8192 x 1024]
    __bf16* Kp  = Qp + Pe;                   // [8192 x 1024]
    __bf16* Vt  = Kp + Pe;                   // [1024 x 8192] (transposed V-projection)
    __bf16* At  = Vt + Pe;                   // [4 x 2048 x 2048]

    const dim3 blk(256);
    const float scale = 0.03125f;            // 1/sqrt(1024)

    // --- pre-pass: bf16 conversion + weight transposes ---
    cvt_f32_bf16<<<dim3((unsigned)(Pe / 4 / 256)), blk, 0, stream>>>(q, qb, Pe);
    cvt_f32_bf16<<<dim3((unsigned)(Pe / 4 / 256)), blk, 0, stream>>>(k, kb, Pe);
    cvt_f32_bf16<<<dim3((unsigned)(Pe / 4 / 256)), blk, 0, stream>>>(v, vb, Pe);
    transpose_cvt<<<dim3(Dn / 32, Dn / 32), blk, 0, stream>>>(Wq, Wqt, Dn, Dn);
    transpose_cvt<<<dim3(Dn / 32, Dn / 32), blk, 0, stream>>>(Wk, Wkt, Dn, Dn);
    transpose_cvt<<<dim3(Dn / 32, Dn / 32), blk, 0, stream>>>(Wv, Wvt, Dn, Dn);

    // --- QKV projections (Q,K normal; V stored transposed for the AV GEMM) ---
    const dim3 gp(Dn / BN, Mp / BM, 1);
    wmma_gemm<true, true, false><<<gp, blk, 0, stream>>>(
        qb, Wqt, bq, Qp, Dn, Dn, Dn, Dn, 0, 0, 0, 1.0f);
    wmma_gemm<true, true, false><<<gp, blk, 0, stream>>>(
        kb, Wkt, bk, Kp, Dn, Dn, Dn, Dn, 0, 0, 0, 1.0f);
    wmma_gemm<true, true, true><<<gp, blk, 0, stream>>>(
        vb, Wvt, bv, Vt, Dn, Dn, Dn, Mp, 0, 0, 0, 1.0f);

    // --- logits = (Qp @ Kp^T) * scale  -> f32, batched ---
    const dim3 gl(Sn / BN, Sn / BM, Bn);
    wmma_gemm<false, false, false><<<gl, blk, 0, stream>>>(
        Qp, Kp, nullptr, Lg, Dn, Dn, Dn, Sn,
        (long)Sn * Dn, (long)Sn * Dn, (long)Sn * Sn, scale);

    // --- softmax rows -> bf16 attn ---
    softmax_rows<<<dim3(Bn * Sn), blk, 0, stream>>>(Lg, At, Sn);

    // --- out = attn @ V  (B operand = Vt, K-contiguous), f32 out ---
    const dim3 ga(Dn / BN, Sn / BM, Bn);
    wmma_gemm<false, false, false><<<ga, blk, 0, stream>>>(
        At, Vt, nullptr, out, Sn, Sn, Mp, Dn,
        (long)Sn * Sn, (long)Sn, (long)Sn * Dn, 1.0f);
}
